// BMNISTAutoEncoder_37666863186278
// MI455X (gfx1250) — matrix operand order repaired and checked by reading, not compiled, not run on hardware
//
#include <hip/hip_runtime.h>

// ---------------------------------------------------------------------------
// MI455X / gfx1250 implementation. All GEMM-shaped work (conv included) goes
// through v_wmma_f32_16x16x32_bf16 (f32 accum). Cubic resamplers are VALU.
// ---------------------------------------------------------------------------

typedef __attribute__((ext_vector_type(16))) __bf16 v16bf;
typedef __attribute__((ext_vector_type(8)))  float  v8f;

union Frag { uint4 q[2]; v16bf v; };

#define BT_TOT 8192     // B*T
#define BATCH  512
#define TSTEP  16
#define FS     64
#define DG     28
#define CO     37       // conv out 37x37
#define FC     1369     // 37*37
#define FCP    1376     // padded to mult of 32

// Frame LDS row stride: 104 elems => 208B rows: 16B-aligned b128 loads and
// conflict-free banks (52 dwords/row, gcd(52,64)=4, period 16).
#define FROW   104

// ---------------------------------------------------------------------------
// Weight transpose + f32->bf16 convert:  WT[n][k] = W[k][n], zero padded.
// ---------------------------------------------------------------------------
__global__ void wconv_kernel(const float* __restrict__ W, __bf16* __restrict__ WT,
                             int K, int N, int Kpad, int Npad) {
  int i = blockIdx.x * 256 + threadIdx.x;
  if (i >= Kpad * Npad) return;
  int n = i / Kpad, k = i % Kpad;
  WT[i] = (k < K && n < N) ? (__bf16)W[(size_t)k * N + n] : (__bf16)0.0f;
}

// ---------------------------------------------------------------------------
// Banded conv "B" table:  Bg[ky][n][c] = kflip[ky][c-n] if 0<=c-n<28 else 0
// kflip[ky][kx] = digit_mean[27-ky][27-kx] (true convolution flips the kernel)
// Layout [28][16][64] bf16 (56 KB, L2-resident).
// ---------------------------------------------------------------------------
__global__ void build_convB_kernel(const float* __restrict__ dm, __bf16* __restrict__ Bg) {
  int i = blockIdx.x * 256 + threadIdx.x;
  if (i >= 28 * 16 * 64) return;
  int ky = i >> 10, j = (i >> 6) & 15, c = i & 63;
  int d = c - j;
  float wv = (d >= 0 && d < DG) ? dm[(27 - ky) * DG + (27 - d)] : 0.0f;
  Bg[i] = (__bf16)wv;
}

// ---------------------------------------------------------------------------
// Fused conv(64x64 * 28x28 valid) + softmax(1369) -> bf16 row [1376].
// One block per frame, 9 waves, one 16x16 output tile per wave.
// out[y0+i][x0+j] = sum_ky sum_c frame[y0+i+ky][x0+c] * Bg[ky][j][c]
// ---------------------------------------------------------------------------
__global__ __launch_bounds__(288) void conv_softmax_kernel(
    const float* __restrict__ frames, const __bf16* __restrict__ Bg,
    __bf16* __restrict__ Asm) {
  __shared__ __align__(16) __bf16 fsh[80 * FROW];   // zero-padded bf16 frame
  __shared__ float fcsh[FCP];
  __shared__ float red[320];
  const int tid = threadIdx.x;
  const int bt  = blockIdx.x;

  for (int i = tid; i < 80 * FROW; i += 288) fsh[i] = (__bf16)0.0f;
  for (int i = tid; i < FCP; i += 288) fcsh[i] = 0.0f;
  __syncthreads();
  const float* fr = frames + (size_t)bt * (FS * FS);
  for (int i = tid; i < FS * FS; i += 288) {
    int y = i >> 6, x = i & 63;
    fsh[y * FROW + x] = (__bf16)fr[i];
  }
  __syncthreads();

  const int wid = tid >> 5, lane = tid & 31;
  const int ty = wid / 3, tx = wid % 3;
  const int y0 = ty * 16, x0 = tx * 16;
  const int ml = lane & 15, hi = lane >> 4;

  v8f acc = 0.0f;
  const __bf16* bbase = Bg + ml * 64 + hi * 8;    // B fragment: lanes0-15 K0..7, hi lanes K8..15 (same as A)
  for (int ky = 0; ky < DG; ++ky) {
    const __bf16* ap = &fsh[(y0 + ml + ky) * FROW + x0 + hi * 8]; // A: lanes0-15 K0..7, hi lanes K8..15
    const __bf16* bp = bbase + ky * 1024;
#pragma unroll
    for (int kh = 0; kh < 2; ++kh) {
      const int kc = kh * 32;
      Frag a, b;
      a.q[0] = *(const uint4*)(ap + kc);      a.q[1] = *(const uint4*)(ap + kc + 16);
      b.q[0] = *(const uint4*)(bp + kc);      b.q[1] = *(const uint4*)(bp + kc + 16);
      acc = __builtin_amdgcn_wmma_f32_16x16x32_bf16(false, a.v, false, b.v,
                                                    (short)0, acc, false, false);
    }
  }
#pragma unroll
  for (int r = 0; r < 8; ++r) {
    int y = y0 + r + hi * 8, x = x0 + ml;
    if (y < CO && x < CO) fcsh[y * CO + x] = acc[r];
  }
  __syncthreads();

  // softmax over fcsh[0..1368]
  float mx = -1e30f;
  for (int i = tid; i < FC; i += 288) mx = fmaxf(mx, fcsh[i]);
  red[tid] = mx; __syncthreads();
  if (tid < 32) { float m = red[tid]; for (int i = tid + 32; i < 288; i += 32) m = fmaxf(m, red[i]); red[tid] = m; }
  __syncthreads();
  if (tid == 0) { float m = red[0]; for (int i = 1; i < 32; ++i) m = fmaxf(m, red[i]); red[288] = m; }
  __syncthreads();
  mx = red[288];
  float s = 0.0f;
  for (int i = tid; i < FC; i += 288) { float e = __expf(fcsh[i] - mx); fcsh[i] = e; s += e; }
  red[tid] = s; __syncthreads();
  if (tid < 32) { float v = red[tid]; for (int i = tid + 32; i < 288; i += 32) v += red[i]; red[tid] = v; }
  __syncthreads();
  if (tid == 0) { float v = red[0]; for (int i = 1; i < 32; ++i) v += red[i]; red[289] = v; }
  __syncthreads();
  const float inv = 1.0f / red[289];
  __bf16* out = Asm + (size_t)bt * FCP;
  for (int i = tid; i < FCP; i += 288)
    out[i] = (__bf16)(i < FC ? fcsh[i] * inv : 0.0f);
}

// ---------------------------------------------------------------------------
// Generic WMMA GEMM: D = act(A[M,Kpad] * BT[Npad,Kpad]^T + bias)
// One wave per 16x16 tile. grid = (M/16, Npad/16), block = 32.
// modes: 0 relu->bf16(padded)  1 relu->f32  2 tanh->f32  3 sigmoid->f32
//        4 linear->bf16(padded)
// ---------------------------------------------------------------------------
__global__ __launch_bounds__(32) void gemm_wmma_kernel(
    const __bf16* __restrict__ A, const __bf16* __restrict__ BT,
    const float* __restrict__ bias, void* __restrict__ Cout,
    int Nreal, int Kpad, int ldc, int mode) {
  const int mt = blockIdx.x, nt = blockIdx.y, lane = threadIdx.x;
  const int ml = lane & 15, hi = lane >> 4;
  const __bf16* Ap = A  + (size_t)(mt * 16 + ml) * Kpad + hi * 8;
  const __bf16* Bp = BT + (size_t)(nt * 16 + ml) * Kpad + hi * 8;
  v8f acc = 0.0f;
  for (int k = 0; k < Kpad; k += 32) {
    Frag a, b;
    a.q[0] = *(const uint4*)(Ap + k);  a.q[1] = *(const uint4*)(Ap + k + 16);
    b.q[0] = *(const uint4*)(Bp + k);  b.q[1] = *(const uint4*)(Bp + k + 16);
    acc = __builtin_amdgcn_wmma_f32_16x16x32_bf16(false, a.v, false, b.v,
                                                  (short)0, acc, false, false);
  }
  const int nn = nt * 16 + ml;
  const float bv = (nn < Nreal) ? bias[nn] : 0.0f;
#pragma unroll
  for (int r = 0; r < 8; ++r) {
    const int mm = mt * 16 + r + hi * 8;
    float v = acc[r] + bv;
    if (mode == 0) {
      v = fmaxf(v, 0.0f);
      ((__bf16*)Cout)[(size_t)mm * ldc + nn] = (__bf16)(nn < Nreal ? v : 0.0f);
    } else if (mode == 1) {
      if (nn < Nreal) ((float*)Cout)[(size_t)mm * ldc + nn] = fmaxf(v, 0.0f);
    } else if (mode == 2) {
      if (nn < Nreal) ((float*)Cout)[(size_t)mm * ldc + nn] = tanhf(v);
    } else if (mode == 3) {
      if (nn < Nreal) ((float*)Cout)[(size_t)mm * ldc + nn] = 1.0f / (1.0f + __expf(-v));
    } else {
      ((__bf16*)Cout)[(size_t)mm * ldc + nn] = (__bf16)(nn < Nreal ? v : 0.0f);
    }
  }
}

// ---------------------------------------------------------------------------
// Catmull-Rom cubic taps matching jax scale_and_translate (scale=1,
// antialias=False): sample s = o - translate; 4 taps; normalize over
// in-range taps; zero the column if s outside [-0.5, insz-0.5].
// ---------------------------------------------------------------------------
__device__ __forceinline__ void cubic_taps(float s, int insz, int& i0, float w[4]) {
  i0 = (int)floorf(s) - 1;
  float tot = 0.0f;
#pragma unroll
  for (int k = 0; k < 4; ++k) {
    float x = fabsf(s - (float)(i0 + k));
    float wk;
    if (x <= 1.0f)      wk = ((1.5f * x - 2.5f) * x) * x + 1.0f;
    else if (x < 2.0f)  wk = ((-0.5f * x + 2.5f) * x - 4.0f) * x + 2.0f;
    else                wk = 0.0f;
    int idx = i0 + k;
    if (idx < 0 || idx >= insz) wk = 0.0f;
    w[k] = wk; tot += wk;
  }
  const bool oob = (s < -0.5f) || (s > (float)insz - 0.5f);
  const float inv = (!oob && fabsf(tot) > 1e-8f) ? 1.0f / tot : 0.0f;
#pragma unroll
  for (int k = 0; k < 4; ++k) w[k] *= inv;
}

__device__ __forceinline__ int clampi(int v, int lo, int hi) {
  return v < lo ? lo : (v > hi ? hi : v);
}

// crop_frames: 64x64 -> 28x28 per (b,t); write bf16 padded [8192][800]
__global__ __launch_bounds__(256) void crop_kernel(
    const float* __restrict__ frames, const float* __restrict__ zw,
    __bf16* __restrict__ digits) {
  __shared__ float fsh[FS * FS];
  const int bt = blockIdx.x, tid = threadIdx.x;
  const float* fr = frames + (size_t)bt * (FS * FS);
  for (int i = tid; i < FS * FS; i += 256) fsh[i] = fr[i];
  __syncthreads();
  const float tr = 18.0f * (zw[bt * 2 + 1] + 1.0f);  // translate = 36*(w[::-1]+1)/2
  const float tc = 18.0f * (zw[bt * 2 + 0] + 1.0f);
  __bf16* out = digits + (size_t)bt * 800;
  for (int i = tid; i < 800; i += 256) {
    float v = 0.0f;
    if (i < DG * DG) {
      const int r = i / DG, c = i % DG;
      int ir0, ic0; float wr[4], wc[4];
      cubic_taps((float)r - tr, FS, ir0, wr);
      cubic_taps((float)c - tc, FS, ic0, wc);
#pragma unroll
      for (int a = 0; a < 4; ++a) {
        const int ia = clampi(ir0 + a, 0, FS - 1);
        const float wa = wr[a];
#pragma unroll
        for (int bb = 0; bb < 4; ++bb) {
          const int ib = clampi(ic0 + bb, 0, FS - 1);
          v += wa * wc[bb] * fsh[ia * FS + ib];
        }
      }
    }
    out[i] = (__bf16)v;
  }
}

// sum over T and pad: zin[b][n] = sum_t a2[(b*16+t)][n] (bf16, [512][224])
__global__ void reduce_time_kernel(const float* __restrict__ a2, __bf16* __restrict__ zin) {
  int idx = blockIdx.x * 256 + threadIdx.x;
  if (idx >= BATCH * 224) return;
  int b = idx / 224, n = idx % 224;
  float s = 0.0f;
  if (n < 200)
    for (int t = 0; t < TSTEP; ++t) s += a2[((size_t)b * TSTEP + t) * 200 + n];
  zin[idx] = (__bf16)s;
}

// embed_digits: 28x28 -> 64x64 per (b,t) into d_out
__global__ __launch_bounds__(256) void embed_kernel(
    const float* __restrict__ drec, const float* __restrict__ zw,
    float* __restrict__ out) {
  __shared__ float dsh[DG * DG];
  const int bt = blockIdx.x, tid = threadIdx.x;
  const int b = bt >> 4;
  const float* dr = drec + (size_t)b * (DG * DG);
  for (int i = tid; i < DG * DG; i += 256) dsh[i] = dr[i];
  __syncthreads();
  const float tr = 18.0f * (zw[bt * 2 + 1] + 1.0f);  // |28-64| = 36 as well
  const float tc = 18.0f * (zw[bt * 2 + 0] + 1.0f);
  float* o = out + (size_t)bt * (FS * FS);
  for (int i = tid; i < FS * FS; i += 256) {
    const int r = i >> 6, c = i & 63;
    int ir0, ic0; float wr[4], wc[4];
    cubic_taps((float)r - tr, DG, ir0, wr);
    cubic_taps((float)c - tc, DG, ic0, wc);
    float v = 0.0f;
#pragma unroll
    for (int a = 0; a < 4; ++a) {
      const int ia = clampi(ir0 + a, 0, DG - 1);
      const float wa = wr[a];
#pragma unroll
      for (int bb = 0; bb < 4; ++bb) {
        const int ib = clampi(ic0 + bb, 0, DG - 1);
        v += wa * wc[bb] * dsh[ia * DG + ib];
      }
    }
    o[i] = v;
  }
}

// ---------------------------------------------------------------------------
extern "C" void kernel_launch(void* const* d_in, const int* in_sizes, int n_in,
                              void* d_out, int out_size, void* d_ws, size_t ws_size,
                              hipStream_t stream) {
  const float* frames     = (const float*)d_in[0];
  const float* digit_mean = (const float*)d_in[1];
  const float* where_w1   = (const float*)d_in[2];
  const float* where_b1   = (const float*)d_in[3];
  const float* where_w2   = (const float*)d_in[4];
  const float* where_b2   = (const float*)d_in[5];
  const float* where_wloc = (const float*)d_in[6];
  const float* where_bloc = (const float*)d_in[7];
  const float* what_w1    = (const float*)d_in[8];
  const float* what_b1    = (const float*)d_in[9];
  const float* what_w2    = (const float*)d_in[10];
  const float* what_b2    = (const float*)d_in[11];
  const float* what_wloc  = (const float*)d_in[12];
  const float* what_bloc  = (const float*)d_in[13];
  const float* dec_w1     = (const float*)d_in[14];
  const float* dec_b1     = (const float*)d_in[15];
  const float* dec_w2     = (const float*)d_in[16];
  const float* dec_b2     = (const float*)d_in[17];
  const float* dec_w3     = (const float*)d_in[18];
  const float* dec_b3     = (const float*)d_in[19];

  size_t off = 0;
  auto alloc = [&](size_t bytes) -> char* {
    char* p = (char*)d_ws + off;
    off += (bytes + 255) & ~(size_t)255;
    return p;
  };

  // Transposed bf16 weights [Npad][Kpad]
  __bf16* wt1 = (__bf16*)alloc((size_t)224 * 1376 * 2); // where_w1  K1369 N200
  __bf16* wt2 = (__bf16*)alloc((size_t)128 * 224  * 2); // where_w2  K200  N200 (need 100)
  __bf16* wt3 = (__bf16*)alloc((size_t)16  * 128  * 2); // where_wloc K100 N2
  __bf16* wt4 = (__bf16*)alloc((size_t)416 * 800  * 2); // what_w1   K784  N400
  __bf16* wt5 = (__bf16*)alloc((size_t)224 * 416  * 2); // what_w2   K400  N200
  __bf16* wt6 = (__bf16*)alloc((size_t)32  * 224  * 2); // what_wloc K200  N10
  __bf16* wt7 = (__bf16*)alloc((size_t)224 * 32   * 2); // dec_w1    K10   N200
  __bf16* wt8 = (__bf16*)alloc((size_t)416 * 224  * 2); // dec_w2    K200  N400
  __bf16* wt9 = (__bf16*)alloc((size_t)784 * 416  * 2); // dec_w3    K400  N784
  __bf16* Bg  = (__bf16*)alloc((size_t)28 * 16 * 64 * 2); // banded conv table

  // Activations
  __bf16* Asm   = (__bf16*)alloc((size_t)BT_TOT * FCP * 2); // softmax(conv) bf16
  __bf16* h1    = (__bf16*)alloc((size_t)BT_TOT * 224 * 2);
  __bf16* h2    = (__bf16*)alloc((size_t)BT_TOT * 128 * 2);
  float*  zwf   = (float*)alloc((size_t)BT_TOT * 2 * 4);
  __bf16* digb  = (__bf16*)alloc((size_t)BT_TOT * 800 * 2);
  __bf16* w1b   = (__bf16*)alloc((size_t)BT_TOT * 416 * 2);
  float*  a2f   = (float*)alloc((size_t)BT_TOT * 200 * 4);
  __bf16* zin   = (__bf16*)alloc((size_t)BATCH * 224 * 2);
  __bf16* zwhat = (__bf16*)alloc((size_t)BATCH * 32 * 2);
  __bf16* d1b   = (__bf16*)alloc((size_t)BATCH * 224 * 2);
  __bf16* d2b   = (__bf16*)alloc((size_t)BATCH * 416 * 2);
  float*  drec  = (float*)alloc((size_t)BATCH * 784 * 4);

  auto wlaunch = [&](const float* W, __bf16* WT, int K, int N, int Kp, int Np) {
    int tot = Kp * Np;
    wconv_kernel<<<(tot + 255) / 256, 256, 0, stream>>>(W, WT, K, N, Kp, Np);
  };
  wlaunch(where_w1,   wt1, 1369, 200, 1376, 224);
  wlaunch(where_w2,   wt2, 200,  200, 224,  128);
  wlaunch(where_wloc, wt3, 100,  2,   128,  16);
  wlaunch(what_w1,    wt4, 784,  400, 800,  416);
  wlaunch(what_w2,    wt5, 400,  200, 416,  224);
  wlaunch(what_wloc,  wt6, 200,  10,  224,  32);
  wlaunch(dec_w1,     wt7, 10,   200, 32,   224);
  wlaunch(dec_w2,     wt8, 200,  400, 224,  416);
  wlaunch(dec_w3,     wt9, 400,  784, 416,  784);
  build_convB_kernel<<<(28 * 16 * 64 + 255) / 256, 256, 0, stream>>>(digit_mean, Bg);

  // conv + softmax (WMMA)
  conv_softmax_kernel<<<BT_TOT, 288, 0, stream>>>(frames, Bg, Asm);

  // where MLP
  gemm_wmma_kernel<<<dim3(BT_TOT / 16, 14), 32, 0, stream>>>(Asm, wt1, where_b1, h1, 200, 1376, 224, 0);
  gemm_wmma_kernel<<<dim3(BT_TOT / 16, 8),  32, 0, stream>>>(h1,  wt2, where_b2, h2, 100, 224,  128, 0);
  gemm_wmma_kernel<<<dim3(BT_TOT / 16, 1),  32, 0, stream>>>(h2,  wt3, where_bloc, zwf, 2, 128, 2, 2);

  // crop 64->28 (cubic)
  crop_kernel<<<BT_TOT, 256, 0, stream>>>(frames, zwf, digb);

  // what MLP + time reduction
  gemm_wmma_kernel<<<dim3(BT_TOT / 16, 26), 32, 0, stream>>>(digb, wt4, what_b1, w1b, 400, 800, 416, 0);
  gemm_wmma_kernel<<<dim3(BT_TOT / 16, 14), 32, 0, stream>>>(w1b,  wt5, what_b2, a2f, 200, 416, 200, 1);
  reduce_time_kernel<<<(BATCH * 224 + 255) / 256, 256, 0, stream>>>(a2f, zin);
  gemm_wmma_kernel<<<dim3(BATCH / 16, 2),  32, 0, stream>>>(zin, wt6, what_bloc, zwhat, 10, 224, 32, 4);

  // decoder
  gemm_wmma_kernel<<<dim3(BATCH / 16, 14), 32, 0, stream>>>(zwhat, wt7, dec_b1, d1b, 200, 32,  224, 0);
  gemm_wmma_kernel<<<dim3(BATCH / 16, 26), 32, 0, stream>>>(d1b,   wt8, dec_b2, d2b, 400, 224, 416, 0);
  gemm_wmma_kernel<<<dim3(BATCH / 16, 49), 32, 0, stream>>>(d2b,   wt9, dec_b3, drec, 784, 416, 784, 3);

  // embed 28->64 (cubic) -> output
  embed_kernel<<<BT_TOT, 256, 0, stream>>>(drec, zwf, (float*)d_out);

  (void)in_sizes; (void)n_in; (void)out_size; (void)ws_size;
}
